// SelfAttention_21861383536767
// MI455X (gfx1250) — matrix operand order repaired and checked by reading, not compiled, not run on hardware
//
#include <hip/hip_runtime.h>
#include <hip/hip_bf16.h>

// ---------------------------------------------------------------------------
// Self-attention block for MI455X (gfx1250), bf16 WMMA + f32 accumulate,
// async global->LDS (ASYNCcnt) double-buffered tile pipeline.
// B=2, T=2048, E=1024, H=16, D=64.  Workspace use: 64 MB.
// ---------------------------------------------------------------------------

#define EMB   1024
#define HEADS 16
#define HDIM  64
#define BATCH 2
#define SEQ   2048
#define MROWS (BATCH * SEQ)            // 4096
#define INV_SCALE 0.17677669529663687f // 1 / 1024^0.25

typedef __attribute__((ext_vector_type(16))) __bf16 bf16x16;
typedef __attribute__((ext_vector_type(8)))  float  f32x8;

struct Q2 { uint4 lo, hi; };

static __device__ __forceinline__ bf16x16 mkfrag(uint4 lo, uint4 hi) {
    Q2 q; q.lo = lo; q.hi = hi;
    return __builtin_bit_cast(bf16x16, q);
}

// f32 -> bf16 (round to nearest even), stored as ushort
static __device__ __forceinline__ unsigned short f2bf(float f) {
    unsigned int u = __builtin_bit_cast(unsigned int, f);
    u += 0x7FFFu + ((u >> 16) & 1u);
    return (unsigned short)(u >> 16);
}

// LDS byte offset of a __shared__ object (generic addr low 32 bits = LDS offset)
static __device__ __forceinline__ unsigned lds_off(const void* p) {
    return (unsigned)(unsigned long long)p;
}

// async global -> LDS, 16 bytes per lane, tracked by ASYNCcnt
static __device__ __forceinline__ void async_ld_b128(unsigned lds, const void* gptr) {
    asm volatile("global_load_async_to_lds_b128 %0, %1, off"
                 :: "v"(lds), "v"((unsigned long long)gptr)
                 : "memory");
}

static __device__ __forceinline__ void wait_async0() {
    asm volatile("s_wait_asynccnt 0x0" ::: "memory");
}

// ---------------------------------------------------------------------------
// Elementwise f32 -> bf16 conversion
// ---------------------------------------------------------------------------
__global__ __launch_bounds__(256) void cvt_bf16_kernel(const float* __restrict__ src,
                                                       unsigned short* __restrict__ dst,
                                                       int n) {
    int i = blockIdx.x * 256 + threadIdx.x;
    if (i < n) dst[i] = f2bf(src[i]);
}

// ---------------------------------------------------------------------------
// bf16 GEMM: C[M,N](f32) = A[M,K](bf16,row-major) * B[N,K](bf16,row-major)^T
// Block tile 128x128, 8 waves (2 M x 4 N), wave tile 64x32, K-step 32.
// Double-buffered LDS tiles filled by async global->LDS DMA.
// ---------------------------------------------------------------------------
__global__ __launch_bounds__(256) void gemm_bf16_kernel(const unsigned short* __restrict__ A,
                                                        const unsigned short* __restrict__ B,
                                                        float* __restrict__ C,
                                                        const float* __restrict__ bias,
                                                        int useBias, int M, int N, int K) {
    __shared__ __align__(16) unsigned short As[2][128][40];  // pad 8 -> conflict-free frags
    __shared__ __align__(16) unsigned short Bs[2][128][40];  // 40 KB total

    const int tid   = threadIdx.x;
    const int lane  = tid & 31;
    const int wave  = tid >> 5;       // 0..7
    const int waveM = wave >> 2;      // 0..1
    const int waveN = wave & 3;       // 0..3
    const int half  = lane >> 4;
    const int ml    = lane & 15;

    const unsigned short* Ag = A + (size_t)(blockIdx.y * 128) * K;
    const unsigned short* Bg = B + (size_t)(blockIdx.x * 128) * K;

    // this thread's two 16B chunks of each 128x32 tile (512 chunks / 256 thr)
    const int c0 = tid,        r0 = c0 >> 2, p0 = c0 & 3;
    const int c1 = tid + 256,  r1 = c1 >> 2, p1 = c1 & 3;

    f32x8 acc[4][2];
#pragma unroll
    for (int mt = 0; mt < 4; mt++)
#pragma unroll
        for (int nt = 0; nt < 2; nt++) acc[mt][nt] = {};

    // prologue: fill buffer 0
    async_ld_b128(lds_off(&As[0][r0][p0 * 8]), Ag + (size_t)r0 * K + p0 * 8);
    async_ld_b128(lds_off(&As[0][r1][p1 * 8]), Ag + (size_t)r1 * K + p1 * 8);
    async_ld_b128(lds_off(&Bs[0][r0][p0 * 8]), Bg + (size_t)r0 * K + p0 * 8);
    async_ld_b128(lds_off(&Bs[0][r1][p1 * 8]), Bg + (size_t)r1 * K + p1 * 8);
    wait_async0();
    __syncthreads();

    for (int k0 = 0; k0 < K; k0 += 32) {
        const int buf = (k0 >> 5) & 1;
        if (k0 + 32 < K) {  // prefetch next tile into the other buffer
            const int nb = buf ^ 1, kn = k0 + 32;
            async_ld_b128(lds_off(&As[nb][r0][p0 * 8]), Ag + (size_t)r0 * K + kn + p0 * 8);
            async_ld_b128(lds_off(&As[nb][r1][p1 * 8]), Ag + (size_t)r1 * K + kn + p1 * 8);
            async_ld_b128(lds_off(&Bs[nb][r0][p0 * 8]), Bg + (size_t)r0 * K + kn + p0 * 8);
            async_ld_b128(lds_off(&Bs[nb][r1][p1 * 8]), Bg + (size_t)r1 * K + kn + p1 * 8);
        }

        bf16x16 af[4], bfr[2];
#pragma unroll
        for (int mt = 0; mt < 4; mt++) {
            const unsigned short* ap = &As[buf][waveM * 64 + mt * 16 + ml][8 * half];
            af[mt] = mkfrag(*(const uint4*)ap, *(const uint4*)(ap + 16));
        }
#pragma unroll
        for (int nt = 0; nt < 2; nt++) {
            const unsigned short* bp = &Bs[buf][waveN * 32 + nt * 16 + ml][8 * half];
            bfr[nt] = mkfrag(*(const uint4*)bp, *(const uint4*)(bp + 16));
        }
#pragma unroll
        for (int mt = 0; mt < 4; mt++)
#pragma unroll
            for (int nt = 0; nt < 2; nt++)
                acc[mt][nt] = __builtin_amdgcn_wmma_f32_16x16x32_bf16(
                    false, af[mt], false, bfr[nt], (short)0, acc[mt][nt], false, false);

        wait_async0();      // next buffer resident (and this wave's reads retired)
        __syncthreads();
    }

    const int baseM = blockIdx.y * 128 + waveM * 64;
    const int baseN = blockIdx.x * 128 + waveN * 32;
#pragma unroll
    for (int mt = 0; mt < 4; mt++)
#pragma unroll
        for (int nt = 0; nt < 2; nt++) {
            int col  = baseN + nt * 16 + ml;
            float bb = useBias ? bias[col] : 0.0f;
#pragma unroll
            for (int j = 0; j < 8; j++) {
                int row = baseM + mt * 16 + j + 8 * half;
                C[(size_t)row * N + col] = acc[mt][nt][j] + bb;
            }
        }
}

// ---------------------------------------------------------------------------
// Per-head LayerNorm over D=64, fold in 1/E^0.25, pack bf16 as [B,H,T,D].
// One wave per (row, head); 8 waves per block.
// ---------------------------------------------------------------------------
__global__ __launch_bounds__(256) void ln_pack_kernel(const float* __restrict__ raw,
                                                      const float* __restrict__ w,
                                                      const float* __restrict__ bvec,
                                                      unsigned short* __restrict__ out,
                                                      float outscale) {
    const int lane = threadIdx.x & 31;
    const int task = blockIdx.x * 8 + (threadIdx.x >> 5); // 0 .. 65535
    const int row  = task >> 4;                            // 0 .. 4095
    const int h    = task & 15;
    const int b    = row >> 11;
    const int t    = row & 2047;

    const float* src = raw + (size_t)row * EMB + h * HDIM;
    float v0 = src[lane];
    float v1 = src[lane + 32];

    float s = v0 + v1;
#pragma unroll
    for (int m = 16; m > 0; m >>= 1) s += __shfl_xor(s, m, 32);
    float mu = s * (1.0f / 64.0f);
    float d0 = v0 - mu, d1 = v1 - mu;
    float s2 = d0 * d0 + d1 * d1;
#pragma unroll
    for (int m = 16; m > 0; m >>= 1) s2 += __shfl_xor(s2, m, 32);
    float rstd = rsqrtf(s2 * (1.0f / 64.0f) + 1e-5f);

    unsigned short* dst = out + (((size_t)(b * HEADS + h) * SEQ + t) * HDIM);
    dst[lane]      = f2bf((d0 * rstd * w[lane]      + bvec[lane])      * outscale);
    dst[lane + 32] = f2bf((d1 * rstd * w[lane + 32] + bvec[lane + 32]) * outscale);
}

// ---------------------------------------------------------------------------
// Pack V: raw [B,T,H,D] f32 -> bf16 [B,H,D,T]  (transposed for P@V B-fragments)
// ---------------------------------------------------------------------------
__global__ __launch_bounds__(256) void packv_kernel(const float* __restrict__ raw,
                                                    unsigned short* __restrict__ out) {
    int i   = blockIdx.x * 256 + threadIdx.x;   // 0 .. 4194303
    int row = i >> 10, c = i & 1023;
    int b = row >> 11, t = row & 2047;
    int h = c >> 6,   d = c & 63;
    out[(((size_t)(b * HEADS + h) * HDIM + d) * SEQ) + t] = f2bf(raw[i]);
}

// ---------------------------------------------------------------------------
// Flash-style attention.  Grid: (T/64, B*H).  Block: 128 (4 waves).
// Each wave owns 16 q-rows; Q fragments resident; online softmax in registers;
// K/V tiles double-buffered via async global->LDS; P re-laid C->A through LDS.
// Output bf16 [B,T,E].
// ---------------------------------------------------------------------------
__global__ __launch_bounds__(128) void attn_kernel(const unsigned short* __restrict__ Q,
                                                   const unsigned short* __restrict__ K,
                                                   const unsigned short* __restrict__ Vt,
                                                   const float* __restrict__ mask,
                                                   unsigned short* __restrict__ O) {
    __shared__ __align__(16) unsigned short Ks[2][64][72];    // [key][d]
    __shared__ __align__(16) unsigned short Vs[2][64][72];    // [d][key]
    __shared__ __align__(16) unsigned short Ps[4][16][72];    // per-wave P tile

    const int tid  = threadIdx.x;
    const int lane = tid & 31;
    const int wave = tid >> 5;
    const int half = lane >> 4;
    const int ml   = lane & 15;

    const int bh = blockIdx.y;
    const int b  = bh >> 4;
    const int h  = bh & 15;
    const int q0 = blockIdx.x * 64;
    const int qw = q0 + wave * 16;

    const unsigned short* Kg = K  + (size_t)bh * SEQ * HDIM;
    const unsigned short* Vg = Vt + (size_t)bh * HDIM * SEQ;

    // resident Q fragments (M=16, K-dim = d 0..63 -> two 16x32 frags)
    bf16x16 qa[2];
    {
        const unsigned short* qp = Q + ((size_t)bh * SEQ + qw) * HDIM + ml * HDIM;
#pragma unroll
        for (int kk = 0; kk < 2; kk++) {
            uint4 lo = *(const uint4*)(qp + kk * 32 + 8 * half);
            uint4 hi = *(const uint4*)(qp + kk * 32 + 16 + 8 * half);
            qa[kk] = mkfrag(lo, hi);
        }
    }
    float mq[8];
#pragma unroll
    for (int j = 0; j < 8; j++) mq[j] = mask[b * SEQ + qw + j + 8 * half];

    f32x8 o[4];
#pragma unroll
    for (int nt = 0; nt < 4; nt++) o[nt] = {};
    float mrow[8], lrow[8];
#pragma unroll
    for (int j = 0; j < 8; j++) { mrow[j] = -3.0e38f; lrow[j] = 0.0f; }

    // cooperative async fill of one K/V tile pair (64x64 each)
    auto issueKV = [&](int buf, int kb) {
#pragma unroll
        for (int i = 0; i < 4; i++) {
            int c = tid + i * 128;
            int r = c >> 3, p = c & 7;
            async_ld_b128(lds_off(&Ks[buf][r][p * 8]),
                          Kg + ((size_t)(kb * 64 + r)) * HDIM + p * 8);
            async_ld_b128(lds_off(&Vs[buf][r][p * 8]),
                          Vg + (size_t)r * SEQ + kb * 64 + p * 8);
        }
    };

    issueKV(0, 0);
    wait_async0();
    __syncthreads();

    const int NB = SEQ / 64;
    for (int kb = 0; kb < NB; kb++) {
        const int buf = kb & 1;
        if (kb + 1 < NB) issueKV(buf ^ 1, kb + 1);

        // S = Q * K^T  (16 q-rows x 64 keys)
        float sv[4][8];
#pragma unroll
        for (int nt = 0; nt < 4; nt++) {
            f32x8 s = {};
#pragma unroll
            for (int kk = 0; kk < 2; kk++) {
                const unsigned short* kp = &Ks[buf][nt * 16 + ml][kk * 32 + 8 * half];
                bf16x16 bfr = mkfrag(*(const uint4*)kp, *(const uint4*)(kp + 16));
                s = __builtin_amdgcn_wmma_f32_16x16x32_bf16(
                        false, qa[kk], false, bfr, (short)0, s, false, false);
            }
            float mk = mask[b * SEQ + kb * 64 + nt * 16 + ml];
#pragma unroll
            for (int j = 0; j < 8; j++)
                sv[nt][j] = s[j] + __logf(mq[j] * mk + 1e-45f);
        }

        // online softmax per row; write P (bf16) to this wave's LDS tile
#pragma unroll
        for (int j = 0; j < 8; j++) {
            float r = fmaxf(fmaxf(sv[0][j], sv[1][j]), fmaxf(sv[2][j], sv[3][j]));
#pragma unroll
            for (int m = 8; m > 0; m >>= 1) r = fmaxf(r, __shfl_xor(r, m, 32));
            float mn   = fmaxf(mrow[j], r);
            float corr = __expf(mrow[j] - mn);
            float rs = 0.0f;
#pragma unroll
            for (int nt = 0; nt < 4; nt++) {
                float p = __expf(sv[nt][j] - mn);
                sv[nt][j] = p;
                rs += p;
            }
#pragma unroll
            for (int m = 8; m > 0; m >>= 1) rs += __shfl_xor(rs, m, 32);
            lrow[j] = lrow[j] * corr + rs;
            mrow[j] = mn;
#pragma unroll
            for (int nt = 0; nt < 4; nt++) o[nt][j] *= corr;
            int prow = j + 8 * half;
#pragma unroll
            for (int nt = 0; nt < 4; nt++)
                Ps[wave][prow][nt * 16 + ml] = f2bf(sv[nt][j]);
        }
        // no barrier: Ps is private to this wave; per-wave DS ordering suffices

        // O += P * V   (A = P 16x64(keys), B = Vt keys x d)
#pragma unroll
        for (int kk = 0; kk < 2; kk++) {
            const unsigned short* pp = &Ps[wave][ml][kk * 32 + 8 * half];
            bf16x16 pa = mkfrag(*(const uint4*)pp, *(const uint4*)(pp + 16));
#pragma unroll
            for (int nt = 0; nt < 4; nt++) {
                const unsigned short* vp = &Vs[buf][nt * 16 + ml][kk * 32 + 8 * half];
                bf16x16 vb = mkfrag(*(const uint4*)vp, *(const uint4*)(vp + 16));
                o[nt] = __builtin_amdgcn_wmma_f32_16x16x32_bf16(
                            false, pa, false, vb, (short)0, o[nt], false, false);
            }
        }

        wait_async0();      // next K/V buffer resident
        __syncthreads();    // all waves done reading current buffer
    }

    // epilogue: normalize and write bf16 [B,T,E]
#pragma unroll
    for (int j = 0; j < 8; j++) {
        float inv = 1.0f / lrow[j];
        int t = qw + j + 8 * half;
        unsigned short* dst = O + ((size_t)(b * SEQ + t)) * EMB + h * HDIM;
#pragma unroll
        for (int nt = 0; nt < 4; nt++)
            dst[nt * 16 + ml] = f2bf(o[nt][j] * inv);
    }
}

// ---------------------------------------------------------------------------
// Host launcher
// ---------------------------------------------------------------------------
extern "C" void kernel_launch(void* const* d_in, const int* in_sizes, int n_in,
                              void* d_out, int out_size, void* d_ws, size_t ws_size,
                              hipStream_t stream) {
    (void)in_sizes; (void)n_in; (void)out_size; (void)ws_size;

    const float* x     = (const float*)d_in[0];
    const float* mask  = (const float*)d_in[1];
    const float* Wk    = (const float*)d_in[2];
    const float* Wq    = (const float*)d_in[3];
    const float* Wv    = (const float*)d_in[4];
    const float* Wu    = (const float*)d_in[5];
    const float* bu    = (const float*)d_in[6];
    const float* klnw  = (const float*)d_in[7];
    const float* klnb  = (const float*)d_in[8];
    const float* qlnw  = (const float*)d_in[9];
    const float* qlnb  = (const float*)d_in[10];
    float* out = (float*)d_out;

    char* ws = (char*)d_ws;
    const size_t MB = 1u << 20;
    unsigned short* xb  = (unsigned short*)(ws);             //  8 MB  x bf16
    unsigned short* wkb = (unsigned short*)(ws +  8 * MB);   //  2 MB
    unsigned short* wqb = (unsigned short*)(ws + 10 * MB);   //  2 MB
    unsigned short* wvb = (unsigned short*)(ws + 12 * MB);   //  2 MB
    unsigned short* wub = (unsigned short*)(ws + 14 * MB);   //  2 MB
    unsigned short* Kb  = (unsigned short*)(ws + 16 * MB);   //  8 MB  [B,H,T,D]
    unsigned short* Qb  = (unsigned short*)(ws + 24 * MB);   //  8 MB  [B,H,T,D]
    unsigned short* Vtb = (unsigned short*)(ws + 32 * MB);   //  8 MB  [B,H,D,T]
    float*          raw = (float*)(ws + 40 * MB);            // 16 MB  f32 scratch
    unsigned short* Ob  = (unsigned short*)(ws + 56 * MB);   //  8 MB  [B,T,E]

    const int nX = MROWS * EMB;      // 4194304
    const int nW = EMB * EMB;        // 1048576

    cvt_bf16_kernel<<<(nX + 255) / 256, 256, 0, stream>>>(x,  xb,  nX);
    cvt_bf16_kernel<<<(nW + 255) / 256, 256, 0, stream>>>(Wk, wkb, nW);
    cvt_bf16_kernel<<<(nW + 255) / 256, 256, 0, stream>>>(Wq, wqb, nW);
    cvt_bf16_kernel<<<(nW + 255) / 256, 256, 0, stream>>>(Wv, wvb, nW);
    cvt_bf16_kernel<<<(nW + 255) / 256, 256, 0, stream>>>(Wu, wub, nW);

    dim3 ggrid(EMB / 128, MROWS / 128);   // (8, 32)

    // K = LN(x @ Wk^T) * invscale
    gemm_bf16_kernel<<<ggrid, 256, 0, stream>>>(xb, wkb, raw, nullptr, 0, MROWS, EMB, EMB);
    ln_pack_kernel<<<8192, 256, 0, stream>>>(raw, klnw, klnb, Kb, INV_SCALE);
    // Q = LN(x @ Wq^T) * invscale
    gemm_bf16_kernel<<<ggrid, 256, 0, stream>>>(xb, wqb, raw, nullptr, 0, MROWS, EMB, EMB);
    ln_pack_kernel<<<8192, 256, 0, stream>>>(raw, qlnw, qlnb, Qb, INV_SCALE);
    // V = x @ Wv^T, repacked transposed
    gemm_bf16_kernel<<<ggrid, 256, 0, stream>>>(xb, wvb, raw, nullptr, 0, MROWS, EMB, EMB);
    packv_kernel<<<(nX + 255) / 256, 256, 0, stream>>>(raw, Vtb);

    // flash attention
    attn_kernel<<<dim3(SEQ / 64, BATCH * HEADS), 128, 0, stream>>>(Qb, Kb, Vtb, mask, Ob);

    // out = attn_out @ Wu^T + bu
    gemm_bf16_kernel<<<ggrid, 256, 0, stream>>>(Ob, wub, out, bu, 1, MROWS, EMB, EMB);
}
